// FaceConv_6528350290203
// MI455X (gfx1250) — matrix operand order repaired and checked by reading, not compiled, not run on hardware
//
#include <hip/hip_runtime.h>

typedef __attribute__((ext_vector_type(16))) _Float16 v16h;
typedef __attribute__((ext_vector_type(8)))  float    v8f;

#define F_FACES 1000000
#define C_IN    64
#define C_OUT   64
#define KFLAT   256          // C_IN * 4 stacked maps
#define FACES_PER_WAVE  16
#define WAVES_PER_BLOCK 8
#define FACES_PER_BLOCK (FACES_PER_WAVE * WAVES_PER_BLOCK)

union Frag16 { uint4 q[2]; v16h v; _Float16 h[16]; };
union H4     { _Float16 h[4]; uint2 u; };

__global__ __launch_bounds__(256)
void faceconv_wmma_f16(const float* __restrict__ feat,
                       const int*   __restrict__ ckp,
                       const float* __restrict__ W,
                       const float* __restrict__ bias,
                       float*       __restrict__ out)
{
    // B matrix (W) as f16, [n][kk] row-major : 64*256*2 = 32 KB
    __shared__ __align__(16) _Float16 sW[C_OUT * KFLAT];
    // A tiles (feats) as f16, per-wave [m][kk]  : 8*16*256*2 = 64 KB
    __shared__ __align__(16) _Float16 sA[WAVES_PER_BLOCK * FACES_PER_WAVE * KFLAT];

    const int tid  = threadIdx.x;
    const int lane = tid & 31;
    const int wave = tid >> 5;
    const int m    = lane & 15;   // M row / N column owned by this lane
    const int half = lane >> 4;   // lane-group (0: lanes 0-15, 1: lanes 16-31)

    // ---------------- Phase 1a: stage W (fp32 -> f16) into LDS ----------------
    // 16384 elements / 256 threads = 64 each, contiguous float4 loads.
    {
        const float4* Wv = (const float4*)W;
        #pragma unroll
        for (int i = 0; i < 16; ++i) {
            float4 w4 = Wv[tid * 16 + i];
            H4 p;
            p.h[0] = (_Float16)w4.x; p.h[1] = (_Float16)w4.y;
            p.h[2] = (_Float16)w4.z; p.h[3] = (_Float16)w4.w;
            *(uint2*)&sW[tid * 64 + i * 4] = p.u;
        }
    }

    // ---------------- Phase 1b: gather + build feature maps -> LDS A tile ----
    const int tileBase = blockIdx.x * FACES_PER_BLOCK + wave * FACES_PER_WAVE;
    {
        int face = tileBase + m;
        if (face >= F_FACES) face = F_FACES - 1;          // clamp tail (stores masked later)
        const int i0 = ckp[face * 3 + 0];
        const int i1 = ckp[face * 3 + 1];
        const int i2 = ckp[face * 3 + 2];
        const float4* y0p = (const float4*)(feat + (size_t)face * C_IN);
        const float4* n0p = (const float4*)(feat + (size_t)i0   * C_IN);
        const float4* n1p = (const float4*)(feat + (size_t)i1   * C_IN);
        const float4* n2p = (const float4*)(feat + (size_t)i2   * C_IN);
        _Float16* row = &sA[(wave * FACES_PER_WAVE + m) * KFLAT];
        const int cq0 = half * 8;                          // each half-wave covers 32 channels
        #pragma unroll
        for (int j = 0; j < 8; ++j) {
            const int cq = cq0 + j;                        // float4 chunk: channels 4cq..4cq+3
            float4 y  = y0p[cq];
            float4 a0 = n0p[cq];
            float4 a1 = n1p[cq];
            float4 a2 = n2p[cq];
            float ys[4] = { y.x,  y.y,  y.z,  y.w  };
            float s0[4] = { a0.x, a0.y, a0.z, a0.w };
            float s1[4] = { a1.x, a1.y, a1.z, a1.w };
            float s2[4] = { a2.x, a2.y, a2.z, a2.w };
            Frag16 fr;                                     // feats for 4 channels = 16 halves
            #pragma unroll
            for (int c = 0; c < 4; ++c) {
                float su   = s0[c] + s1[c] + s2[c];
                float ring = fabsf(s2[c] - s0[c]) + fabsf(s0[c] - s1[c]) + fabsf(s1[c] - s2[c]);
                float ctr  = fabsf(ys[c] - s0[c]) + fabsf(ys[c] - s1[c]) + fabsf(ys[c] - s2[c]);
                fr.h[c * 4 + 0] = (_Float16)ys[c];
                fr.h[c * 4 + 1] = (_Float16)su;
                fr.h[c * 4 + 2] = (_Float16)ring;
                fr.h[c * 4 + 3] = (_Float16)ctr;
            }
            *(uint4*)&row[cq * 16 + 0] = fr.q[0];          // ds_store_b128 x2
            *(uint4*)&row[cq * 16 + 8] = fr.q[1];
        }
    }

    __syncthreads();

    // ---------------- Phase 2: WMMA GEMM  [16 x 256] x [256 x 64] ------------
    v8f acc[4] = { v8f{0.f}, v8f{0.f}, v8f{0.f}, v8f{0.f} };
    const _Float16* aRow = &sA[(wave * FACES_PER_WAVE + m) * KFLAT];

    #pragma unroll
    for (int ks = 0; ks < 8; ++ks) {
        const int kb = ks * 32;
        // A fragment (16x32 f16): lanes 0-15: K = kb+{0..7, 16..23}; lanes 16-31: +8
        Frag16 a;
        a.q[0] = *(const uint4*)&aRow[kb + half * 8 + 0];
        a.q[1] = *(const uint4*)&aRow[kb + half * 8 + 16];
        #pragma unroll
        for (int nt = 0; nt < 4; ++nt) {
            const int n = nt * 16 + m;
            // B fragment (32x16 f16): lane holds column n, K = kb+{0..7, 16..23}; lanes 16-31: +8
            const _Float16* bp = &sW[n * KFLAT + kb + half * 8];
            Frag16 b;
            b.q[0] = *(const uint4*)&bp[0];
            b.q[1] = *(const uint4*)&bp[16];
            acc[nt] = __builtin_amdgcn_wmma_f32_16x16x32_f16(
                          false, a.v, false, b.v, (short)0, acc[nt], false, false);
        }
    }

    // ---------------- Phase 3: D layout -> bias add -> coalesced stores ------
    #pragma unroll
    for (int nt = 0; nt < 4; ++nt) {
        const int n  = nt * 16 + m;
        const float bv = bias[n];
        #pragma unroll
        for (int r = 0; r < 8; ++r) {
            const int M    = half * 8 + r;                 // C/D VGPR layout
            const int face = tileBase + M;
            if (face < F_FACES)
                out[(size_t)face * C_OUT + n] = acc[nt][r] + bv;
        }
    }
}

extern "C" void kernel_launch(void* const* d_in, const int* in_sizes, int n_in,
                              void* d_out, int out_size, void* d_ws, size_t ws_size,
                              hipStream_t stream) {
    const float* feat = (const float*)d_in[0];   // [F, 64] f32
    const int*   ckp  = (const int*)  d_in[1];   // [F, 3]  i32
    const float* W    = (const float*)d_in[2];   // [64, 64, 4] f32
    const float* bias = (const float*)d_in[3];   // [64] f32
    float*       out  = (float*)d_out;           // [F, 64] f32

    const int nblocks = (F_FACES + FACES_PER_BLOCK - 1) / FACES_PER_BLOCK;
    faceconv_wmma_f16<<<nblocks, 256, 0, stream>>>(feat, ckp, W, bias, out);
}
